// HGPSLPool_10634339025567
// MI455X (gfx1250) — matrix operand order repaired and checked by reading, not compiled, not run on hardware
//
#include <hip/hip_runtime.h>
#include <hip/hip_bf16.h>

typedef __attribute__((ext_vector_type(2)))  float    v2f;
typedef __attribute__((ext_vector_type(8)))  float    v8f;
typedef __attribute__((ext_vector_type(16))) _Float16 v16h;

namespace {
constexpr int   cB   = 64;
constexpr int   cN   = 1024;
constexpr int   cF   = 128;
constexpr int   cEG  = 32768;            // N*DEG
constexpr int   cE   = cB * cEG;         // 2,097,152 edges
constexpr int   cK   = 820;              // ceil(0.8*N)
constexpr int   cP   = cB * cK;          // 52,480
constexpr int   cBN  = cB * cN;          // 65,536
constexpr int   cRT  = (cK + 15) / 16;   // 52 tiles of 16
constexpr float cLAMB = 1.0f;
constexpr float cNEG  = 0.2f;
}

// ---------------- 0: init workspace (zeros + inv = -1) ----------------
__global__ void hgp_init_kernel(float* out_deg, float* in_deg, float* agg, int* inv) {
    size_t i = (size_t)blockIdx.x * blockDim.x + threadIdx.x;
    if (i < (size_t)cBN) { out_deg[i] = 0.0f; in_deg[i] = 0.0f; inv[i] = -1; }
    if (i < (size_t)cBN * cF) agg[i] = 0.0f;
}

// ---------------- 1: degree accumulation ----------------
__global__ void hgp_deg_kernel(const int* srcl, const int* dstl,
                               float* out_deg, float* in_deg) {
    int e = blockIdx.x * blockDim.x + threadIdx.x;
    if (e >= cE) return;
    int b = e / cEG;
    atomicAdd(&out_deg[srcl[e] + b * cN], 1.0f);
    atomicAdd(&in_deg [dstl[e] + b * cN], 1.0f);
}

// ---------------- 2: message scatter: agg[dst] += feat[src]*src_norm*e_feat ----
__global__ void hgp_msg_kernel(const float* feat, const float* e_feat,
                               const int* srcl, const int* dstl,
                               const float* out_deg, float* agg) {
    int t = blockIdx.x * blockDim.x + threadIdx.x;   // E*32 threads
    int e = t >> 5, lane = t & 31;
    if (e >= cE) return;
    int b   = e / cEG;
    int src = srcl[e] + b * cN;
    int dst = dstl[e] + b * cN;
    float scale = e_feat[e] * rsqrtf(fmaxf(out_deg[src], 1.0f));
    const float4 v = *(const float4*)(feat + (size_t)src * cF + lane * 4);
    float* ap = agg + (size_t)dst * cF + lane * 4;
    atomicAdd(ap + 0, v.x * scale);
    atomicAdd(ap + 1, v.y * scale);
    atomicAdd(ap + 2, v.z * scale);
    atomicAdd(ap + 3, v.w * scale);
}

// ---------------- 3: score = sum_f |feat - agg*dst_norm| (one wave per node) ---
__global__ void hgp_score_kernel(const float* feat, const float* agg,
                                 const float* in_deg, float* score) {
    int t = blockIdx.x * blockDim.x + threadIdx.x;
    int n = t >> 5, lane = t & 31;
    if (n >= cBN) return;
    float dn = rsqrtf(fmaxf(in_deg[n], 1.0f));
    const float* fr = feat + (size_t)n * cF;
    const float* ar = agg  + (size_t)n * cF;
    float acc = 0.0f;
    #pragma unroll
    for (int f = lane; f < cF; f += 32) acc += fabsf(fr[f] - ar[f] * dn);
    #pragma unroll
    for (int off = 16; off > 0; off >>= 1) acc += __shfl_xor(acc, off, 32);
    if (lane == 0) score[n] = acc;
}

// ---------------- 4: per-graph top-K via bitonic sort (descending, stable) ----
__global__ __launch_bounds__(1024)
void hgp_topk_kernel(const float* score, int* inv, int* perm_out) {
    __shared__ unsigned long long sk[cN];
    int b = blockIdx.x, tid = threadIdx.x;
    float sc = score[b * cN + tid];          // sc >= 0 -> bits monotone unsigned
    sk[tid] = ((unsigned long long)__float_as_uint(sc) << 32) |
              (unsigned)(~(unsigned)tid);    // ties -> lower index first
    __syncthreads();
    for (unsigned k = 2; k <= (unsigned)cN; k <<= 1) {
        for (unsigned j = k >> 1; j > 0; j >>= 1) {
            unsigned ixj = (unsigned)tid ^ j;
            if (ixj > (unsigned)tid) {
                unsigned long long x = sk[tid], y = sk[ixj];
                bool up = ((tid & k) == 0);                  // overall descending
                if (up ? (x < y) : (x > y)) { sk[tid] = y; sk[ixj] = x; }
            }
            __syncthreads();
        }
    }
    if (tid < cK) {
        int idx   = (int)(~(unsigned)(sk[tid] & 0xffffffffULL));
        int oldid = b * cN + idx;
        perm_out[b * cK + tid] = oldid;
        inv[oldid] = b * cK + tid;
    }
}

// ---------------- 5: feat_p = feat[perm] (one wave per kept node) ----------
__global__ void hgp_gather_kernel(const float* feat, const int* perm, float* featp) {
    int t = blockIdx.x * blockDim.x + threadIdx.x;   // P*32 threads
    int p = t >> 5, lane = t & 31;
    if (p >= cP) return;
    int old = perm[p];
    const float4 v = *(const float4*)(feat + (size_t)old * cF + lane * 4);
    *(float4*)(featp + (size_t)p * cF + lane * 4) = v;
}

// ---------------- 6: [a_src|a_dst] = feat_p @ att via WMMA f16 16x16x32 ------
// One wave per 16-row tile of feat_p. B holds att_src in col0, att_dst in col1.
// All loads unconditional + vectorized; lane masking done arithmetically so
// EXEC stays fully convergent into the WMMA (no divergent load blocks).
__global__ __launch_bounds__(256)
void hgp_attgemv_kernel(const float* featp, const float* att,
                        float* a_src, float* a_dst) {
    int gwave = (blockIdx.x * blockDim.x + threadIdx.x) >> 5;  // 3280 tiles
    int lane  = threadIdx.x & 31;
    int hi    = lane >> 4;       // lane half
    int ll    = lane & 15;
    int i0    = gwave * 16;      // P % 16 == 0 -> always full tiles

    // per-lane att base: lane col 0 -> att_src, col 1 -> att_dst, others masked
    const float* attb = att + ((ll == 1) ? cF : 0);
    const float  bmask = (ll < 2) ? 1.0f : 0.0f;
    const float* arow = featp + (size_t)(i0 + ll) * cF;

    v8f c = {};
    #pragma unroll
    for (int kc = 0; kc < cF / 32; ++kc) {
        // A: 16x32 f16.  M = ll (all lanes); K block split by lane half.
        const float* rp = arow + kc * 32 + hi * 16;       // 64B aligned
        float4 a0 = *(const float4*)(rp + 0);
        float4 a1 = *(const float4*)(rp + 4);
        float4 a2 = *(const float4*)(rp + 8);
        float4 a3 = *(const float4*)(rp + 12);
        v16h a;
        a[0]=(_Float16)a0.x; a[1]=(_Float16)a0.y; a[2]=(_Float16)a0.z; a[3]=(_Float16)a0.w;
        a[4]=(_Float16)a1.x; a[5]=(_Float16)a1.y; a[6]=(_Float16)a1.z; a[7]=(_Float16)a1.w;
        a[8]=(_Float16)a2.x; a[9]=(_Float16)a2.y; a[10]=(_Float16)a2.z; a[11]=(_Float16)a2.w;
        a[12]=(_Float16)a3.x; a[13]=(_Float16)a3.y; a[14]=(_Float16)a3.z; a[15]=(_Float16)a3.w;
        // B: 32x16 f16. N = ll; b[e] = B[hi*16 + e][ll].
        const float* bp = attb + kc * 32 + hi * 16;       // 64B aligned
        float4 b0 = *(const float4*)(bp + 0);
        float4 b1 = *(const float4*)(bp + 4);
        float4 b2 = *(const float4*)(bp + 8);
        float4 b3 = *(const float4*)(bp + 12);
        v16h bm;
        bm[0]=(_Float16)(b0.x*bmask); bm[1]=(_Float16)(b0.y*bmask);
        bm[2]=(_Float16)(b0.z*bmask); bm[3]=(_Float16)(b0.w*bmask);
        bm[4]=(_Float16)(b1.x*bmask); bm[5]=(_Float16)(b1.y*bmask);
        bm[6]=(_Float16)(b1.z*bmask); bm[7]=(_Float16)(b1.w*bmask);
        bm[8]=(_Float16)(b2.x*bmask); bm[9]=(_Float16)(b2.y*bmask);
        bm[10]=(_Float16)(b2.z*bmask); bm[11]=(_Float16)(b2.w*bmask);
        bm[12]=(_Float16)(b3.x*bmask); bm[13]=(_Float16)(b3.y*bmask);
        bm[14]=(_Float16)(b3.z*bmask); bm[15]=(_Float16)(b3.w*bmask);
        c = __builtin_amdgcn_wmma_f32_16x16x32_f16(
                /*neg_a=*/false, a, /*neg_b=*/false, bm,
                /*c_mod=*/(short)0, c, /*reuse_a=*/false, /*reuse_b=*/false);
    }
    // D: vgpr r -> row i0 + r + 8*hi, col = ll
    #pragma unroll
    for (int r = 0; r < 8; ++r) {
        int row = i0 + r + hi * 8;
        if (ll == 0)      a_src[row] = c[r];
        else if (ll == 1) a_dst[row] = c[r];
    }
}

// ---------------- 7: w tiles = lrelu(s_i + d_j) via WMMA f32 16x16x4 ---------
// Rank-1 trick: A = [s | 1 | 0 | 0] (16x4), B = [1^T ; d^T ; 0 ; 0] (4x16)
// -> D[m][n] = s[m] + d[n].  One wave per 16-row strip, loops column tiles.
__global__ __launch_bounds__(256)
void hgp_wtile_kernel(const float* a_src, const float* a_dst, float* e_out) {
    int b    = blockIdx.x;
    int wave = threadIdx.x >> 5;
    int lane = threadIdx.x & 31;
    int rt   = blockIdx.y * 8 + wave;
    if (rt >= cRT) return;
    int i0 = rt * 16;
    int ll = lane & 15;
    int hi = lane >> 4;

    // A layout (32-bit 16x4): lanes 0-15 hold K=0,1 ; lanes 16-31 hold K=2,3.
    int si = i0 + ll; if (si >= cK) si = cK - 1;   // clamp; invalid rows not stored
    float sv = a_src[b * cK + si];
    float hm = hi ? 0.0f : 1.0f;
    v2f a;
    a.x = sv * hm;            // K=0 col: s   (K=2: 0)
    a.y = hm;                 // K=1 col: 1   (K=3: 0)

    float* wB = e_out + (size_t)b * cK * cK;
    for (int jt = 0; jt < cRT; ++jt) {
        int j0 = jt * 16;
        int dj = j0 + ll; if (dj >= cK) dj = cK - 1;
        float dv = a_dst[b * cK + dj];
        // B layout (32-bit 4x16): N = ll ; VGPR0 rows K=0(lo)/K=2(hi), VGPR1 K=1/K=3.
        v2f bb;
        bb.x = hm;                 // row K=0: ones
        bb.y = dv * hm;            // row K=1: d
        v8f c = {};
        c = __builtin_amdgcn_wmma_f32_16x16x4_f32(
                /*neg_a=*/false, a, /*neg_b=*/false, bb,
                /*c_mod=*/(short)0, c, /*reuse_a=*/false, /*reuse_b=*/false);
        // D: vgpr r -> row = i0 + r + 8*hi, col = j0 + ll (two 64B runs per r)
        #pragma unroll
        for (int r = 0; r < 8; ++r) {
            int row = i0 + r + hi * 8;
            int col = j0 + ll;
            if (row < cK && col < cK) {
                float v = c[r];
                v = (v >= 0.0f) ? v : cNEG * v;   // leaky_relu
                wB[(size_t)row * cK + col] = v;
            }
        }
    }
}

// ---------------- 8: scatter-add lamb*e_feat on surviving edges ------------
__global__ void hgp_edgeadd_kernel(const int* srcl, const int* dstl,
                                   const float* e_feat, const int* inv,
                                   float* e_out) {
    int e = blockIdx.x * blockDim.x + threadIdx.x;
    if (e >= cE) return;
    int b   = e / cEG;
    int ns  = inv[srcl[e] + b * cN];
    int nd  = inv[dstl[e] + b * cN];
    if (ns >= 0 && nd >= 0)
        atomicAdd(&e_out[(size_t)ns * cK + (nd % cK)], cLAMB * e_feat[e]);
}

// ---------------- 9: column softmax (axis = src/rows) in-place --------------
__global__ void hgp_softmax_kernel(float* e_out) {
    int b = blockIdx.y;
    int j = blockIdx.x * blockDim.x + threadIdx.x;
    if (j >= cK) return;
    float* wB = e_out + (size_t)b * cK * cK;
    float m = -3.0e38f, s = 0.0f;
    for (int i = 0; i < cK; ++i) {               // online max/sum, coalesced
        float x  = wB[(size_t)i * cK + j];
        float mn = fmaxf(m, x);
        s = s * __expf(m - mn) + __expf(x - mn);
        m = mn;
    }
    float r = 1.0f / s;
    for (int i = 0; i < cK; ++i) {
        float x = wB[(size_t)i * cK + j];
        wB[(size_t)i * cK + j] = __expf(x - m) * r;
    }
}

extern "C" void kernel_launch(void* const* d_in, const int* in_sizes, int n_in,
                              void* d_out, int out_size, void* d_ws, size_t ws_size,
                              hipStream_t stream) {
    const float* feat   = (const float*)d_in[0];   // [B*N, F]
    const float* e_feat = (const float*)d_in[1];   // [B*EG]
    const float* att    = (const float*)d_in[2];   // [2F]
    const int*   srcl   = (const int*)d_in[3];     // [B, EG]
    const int*   dstl   = (const int*)d_in[4];     // [B, EG]

    // Output layout: feat_p [P*F] f32 | e_out [B*K*K] f32 | perm [P] i32
    float* featp = (float*)d_out;
    float* e_out = featp + (size_t)cP * cF;
    int*   permo = (int*)(e_out + (size_t)cB * cK * cK);

    // Workspace layout (~36 MB)
    float* out_deg = (float*)d_ws;
    float* in_deg  = out_deg + cBN;
    float* agg     = in_deg + cBN;
    float* score   = agg + (size_t)cBN * cF;
    int*   inv     = (int*)(score + cBN);
    float* a_src   = (float*)(inv + cBN);
    float* a_dst   = a_src + cP;

    hgp_init_kernel<<<(cBN * cF + 255) / 256, 256, 0, stream>>>(out_deg, in_deg, agg, inv);
    hgp_deg_kernel<<<(cE + 255) / 256, 256, 0, stream>>>(srcl, dstl, out_deg, in_deg);
    hgp_msg_kernel<<<(cE * 32) / 256, 256, 0, stream>>>(feat, e_feat, srcl, dstl, out_deg, agg);
    hgp_score_kernel<<<(cBN * 32) / 256, 256, 0, stream>>>(feat, agg, in_deg, score);
    hgp_topk_kernel<<<cB, cN, 0, stream>>>(score, inv, permo);
    hgp_gather_kernel<<<(cP * 32) / 256, 256, 0, stream>>>(feat, permo, featp);
    hgp_attgemv_kernel<<<((cP / 16) * 32) / 256, 256, 0, stream>>>(featp, att, a_src, a_dst);
    hgp_wtile_kernel<<<dim3(cB, (cRT + 7) / 8), 256, 0, stream>>>(a_src, a_dst, e_out);
    hgp_edgeadd_kernel<<<(cE + 255) / 256, 256, 0, stream>>>(srcl, dstl, e_feat, inv, e_out);
    hgp_softmax_kernel<<<dim3((cK + 255) / 256, cB), 256, 0, stream>>>(e_out);
}
